// DebateModel_73048803770686
// MI455X (gfx1250) — matrix operand order repaired and checked by reading, not compiled, not run on hardware
//
#include <hip/hip_runtime.h>
#include <hip/hip_bf16.h>
#include <math.h>

// ---------------- problem constants (from reference) ----------------
#define C_N   64      // comments (batch)
#define L_N   1024    // tokens per comment
#define A_N   32      // ADUs per comment
#define E_N   48      // edges per comment
#define H_L   80      // LSTM hidden
#define GATE  (4*H_L) // 320
#define SPAN  320     // SPAN_DIM
#define EMB   768
#define HEADS 3
#define HID   256     // HAN_HID
#define ADU   256     // ADU_DIM
#define CTX   200
#define DWO   1608    // SPAN + 5*ADU + 4 + 4
#define DOUT  1808    // CTX + DWO

// ---------------- WMMA types ----------------
typedef __attribute__((ext_vector_type(16))) __bf16 v16bf;
typedef __attribute__((ext_vector_type(8)))  float  v8f;
union BFrag { unsigned short s[16]; uint4 q[2]; v16bf v; };

__device__ inline unsigned short f2bf(float f) {
  unsigned int u = __float_as_uint(f);
  u += 0x7FFFu + ((u >> 16) & 1u);   // round-to-nearest-even
  return (unsigned short)(u >> 16);
}
__device__ inline unsigned long long pack4bf(float4 v) {
  unsigned long long lo = (unsigned long long)(f2bf(v.x) | ((unsigned)f2bf(v.y) << 16));
  unsigned long long hi = (unsigned long long)(f2bf(v.z) | ((unsigned)f2bf(v.w) << 16));
  return lo | (hi << 32);
}
__device__ inline float sigf(float x) { return 1.0f / (1.0f + __expf(-x)); }

// =====================================================================
// Generic GEMM: Y[M,N] = X[M,K] * W' (+bias), fp32 in/out, bf16 WMMA core.
// W_MODE: 0 -> W is [K,N]        (W[k*N+n])
//         1 -> W is [N,K]        (W[n*K+k])
//         2 -> W is grouped-KN:  W[((n>>8)*K + k)*256 + (n&255)]  (W_gat)
//
// Block: 256 threads = 8 waves (4x2). Block tile 64x64, K-step 64.
// Each wave computes a 16x32 slab -> 4 WMMAs per staging barrier.
// LDS tiles are staged in *fragment order* so a fragment load is
// 2x ds_load_b128. Full interior tiles use float4 global loads +
// ds_store_b64 packed stores (no per-element guards).
//
// A fragment layout (16-bit A 16x32, ISA 7.12.2): lane L holds row M=L&15;
// lanes 0-15 hold K in {0..7,16..23}, lanes 16-31 hold K in {8..15,24..31}:
//   element (rr, cc):  grp  = (cc>>3)&1,  slot = (cc&7) + ((cc>>4)&1)*8
// B fragment layout (16-bit B 32x16): lane L holds col N=L&15;
// lanes 0-15 hold K in {0..7,16..23}, lanes 16-31 hold K in {8..15,24..31} -> col-major k runs.
// =====================================================================
template <int W_MODE>
__global__ __launch_bounds__(256)
void gemm_wmma(const float* __restrict__ X, const float* __restrict__ W,
               const float* __restrict__ bias, float* __restrict__ Y,
               int M, int N, int K)
{
  __shared__ __align__(16) unsigned short Af[4 * 2 * 32 * 16]; // [wr][ksub][lane][slot] 8KB
  __shared__ __align__(16) unsigned short Bf[2 * 64 * 32];     // [ksub][ncol][k]        8KB
  const int tid  = threadIdx.x;
  const int m0   = blockIdx.y * 64;
  const int n0   = blockIdx.x * 64;
  const int wave = tid >> 5, lane = tid & 31;
  const int wr   = wave >> 1, wc = wave & 1;
  const bool fullMN = (m0 + 64 <= M) && (n0 + 64 <= N);

  v8f acc0 = {}, acc1 = {};
  for (int k0 = 0; k0 < K; k0 += 64) {
    // prefetch next K-slab of X into cache (global_prefetch_b8 path)
    if (k0 + 64 < K) {
      const int pr = m0 + (tid >> 2), pc = k0 + 64 + (tid & 3) * 16;
      if (pr < M && pc < K) __builtin_prefetch(&X[(size_t)pr * K + pc], 0, 1);
    }
    const bool full = fullMN && (k0 + 64 <= K);
    if (full) {
      // ---- fast path: vectorized, unguarded ----
      // A: 1024 float4 -> packed 8B fragment stores
      for (int i = tid; i < 64 * 16; i += 256) {
        const int r = i >> 4, c4 = (i & 15) * 4;
        const float4 v = *(const float4*)&X[(size_t)(m0 + r) * K + k0 + c4];
        const int ksub = c4 >> 5, cc = c4 & 31;
        const int grp  = (cc >> 3) & 1;
        const int slot = (cc & 7) + ((cc >> 4) & 1) * 8;
        const int base = ((((r >> 4) * 2 + ksub) * 32) + (r & 15) + grp * 16) * 16 + slot;
        *(unsigned long long*)&Af[base] = pack4bf(v);
      }
      // B
      if (W_MODE == 1) {            // coalesce along K of W[N,K]; contiguous frag dest
        for (int i = tid; i < 64 * 16; i += 256) {
          const int nn = i >> 4, k4 = (i & 15) * 4;
          const float4 v = *(const float4*)&W[(size_t)(n0 + nn) * K + k0 + k4];
          const int ksub = k4 >> 5, cc = k4 & 31;
          const int base = ((ksub * 64 + nn) * 32) + (((cc >> 3) & 1) * 16) + (cc & 7) + ((cc >> 4) & 1) * 8;
          *(unsigned long long*)&Bf[base] = pack4bf(v);
        }
      } else {                      // coalesce along N; strided frag dest
        for (int i = tid; i < 64 * 16; i += 256) {
          const int kk = i >> 4, n4 = (i & 15) * 4;
          const int gk = k0 + kk, gn = n0 + n4;
          float4 v;
          if (W_MODE == 0) v = *(const float4*)&W[(size_t)gk * N + gn];
          else             v = *(const float4*)&W[(((size_t)(gn >> 8)) * K + gk) * 256 + (gn & 255)];
          const int ksub = kk >> 5, cc = kk & 31;
          const int di = (((cc >> 3) & 1) * 16) + (cc & 7) + ((cc >> 4) & 1) * 8;
          Bf[((ksub * 64 + n4 + 0) * 32) + di] = f2bf(v.x);
          Bf[((ksub * 64 + n4 + 1) * 32) + di] = f2bf(v.y);
          Bf[((ksub * 64 + n4 + 2) * 32) + di] = f2bf(v.z);
          Bf[((ksub * 64 + n4 + 3) * 32) + di] = f2bf(v.w);
        }
      }
    } else {
      // ---- tail path: guarded scalar, zero-fill OOB ----
      for (int i = tid; i < 64 * 64; i += 256) {
        const int r = i >> 6, c = i & 63;
        const int gm = m0 + r, gk = k0 + c;
        const float v = (gm < M && gk < K) ? X[(size_t)gm * K + gk] : 0.0f;
        const int ksub = c >> 5, cc = c & 31;
        const int grp  = (cc >> 3) & 1;
        const int slot = (cc & 7) + ((cc >> 4) & 1) * 8;
        Af[((((r >> 4) * 2 + ksub) * 32) + (r & 15) + grp * 16) * 16 + slot] = f2bf(v);
      }
      for (int i = tid; i < 64 * 64; i += 256) {
        int kk, nn;
        if (W_MODE == 1) { nn = i >> 6; kk = i & 63; }
        else             { kk = i >> 6; nn = i & 63; }
        const int gk = k0 + kk, gn = n0 + nn;
        float v = 0.0f;
        if (gk < K && gn < N) {
          if (W_MODE == 0)      v = W[(size_t)gk * N + gn];
          else if (W_MODE == 1) v = W[(size_t)gn * K + gk];
          else                  v = W[(((size_t)(gn >> 8)) * K + gk) * 256 + (gn & 255)];
        }
        const int ksub = kk >> 5, cc = kk & 31;
        Bf[((ksub * 64 + nn) * 32) + (((cc >> 3) & 1) * 16) + (cc & 7) + ((cc >> 4) & 1) * 8] = f2bf(v);
      }
    }
    __syncthreads();

    const int colA = wc * 32 + (lane & 15);
    const int kb   = (lane >> 4) * 16;
#pragma unroll
    for (int ks = 0; ks < 2; ++ks) {
      BFrag fa, fb0, fb1;
      { const uint4* p = (const uint4*)&Af[(((wr * 2 + ks) * 32) + lane) * 16];
        fa.q[0] = p[0]; fa.q[1] = p[1]; }
      { const uint4* p = (const uint4*)&Bf[((ks * 64 + colA) * 32) + kb];
        fb0.q[0] = p[0]; fb0.q[1] = p[1]; }
      { const uint4* p = (const uint4*)&Bf[((ks * 64 + colA + 16) * 32) + kb];
        fb1.q[0] = p[0]; fb1.q[1] = p[1]; }
      acc0 = __builtin_amdgcn_wmma_f32_16x16x32_bf16(false, fa.v, false, fb0.v,
                                                     (short)0, acc0, false, false);
      acc1 = __builtin_amdgcn_wmma_f32_16x16x32_bf16(false, fa.v, false, fb1.v,
                                                     (short)0, acc1, false, false);
    }
    __syncthreads();
  }

  // C/D layout: VGPR r -> M = r + 8*(lane>=16), N = lane&15
  const int gn0 = n0 + wc * 32 + (lane & 15);
  const int gn1 = gn0 + 16;
  const int mb  = m0 + wr * 16 + ((lane >> 4) * 8);
  const float bv0 = (bias && gn0 < N) ? bias[gn0] : 0.0f;
  const float bv1 = (bias && gn1 < N) ? bias[gn1] : 0.0f;
#pragma unroll
  for (int r = 0; r < 8; ++r) {
    const int gm = mb + r;
    if (gm < M) {
      if (gn0 < N) Y[(size_t)gm * N + gn0] = acc0[r] + bv0;
      if (gn1 < N) Y[(size_t)gm * N + gn1] = acc1[r] + bv1;
    }
  }
}

// =====================================================================
// LSTM recurrence: one block per sequence (comment) per direction.
// Z = precomputed x@Wih^T + b  (GATE per step). W_hh row cached in regs.
// =====================================================================
__global__ __launch_bounds__(320)
void lstm_seq(const float* __restrict__ Z, const float* __restrict__ Whh,
              float* __restrict__ seq, int reverse, int outOff)
{
  __shared__ float sh_h[H_L];
  __shared__ float sh_z[GATE];
  const int c = blockIdx.x, tid = threadIdx.x;
  float wrow[H_L];
#pragma unroll
  for (int k = 0; k < H_L; ++k) wrow[k] = Whh[tid * H_L + k];
  if (tid < H_L) sh_h[tid] = 0.0f;
  float cc = 0.0f;
  __syncthreads();
  for (int s = 0; s < L_N; ++s) {
    const int t = reverse ? (L_N - 1 - s) : s;
    float z = Z[((size_t)c * L_N + t) * GATE + tid];
#pragma unroll 8
    for (int k = 0; k < H_L; ++k) z += sh_h[k] * wrow[k];
    __syncthreads();
    sh_z[tid] = z;
    __syncthreads();
    if (tid < H_L) {
      float iv = sigf(sh_z[tid]);
      float fv = sigf(sh_z[H_L + tid]);
      float gv = tanhf(sh_z[2 * H_L + tid]);
      float ov = sigf(sh_z[3 * H_L + tid]);
      cc = fv * cc + iv * gv;
      float h = ov * tanhf(cc);
      sh_h[tid] = h;
      seq[((size_t)c * L_N + t) * 160 + outOff + tid] = h;
    }
    __syncthreads();
  }
}

// =====================================================================
// Span representations: cemb (per comment) and masked ADU spans.
// =====================================================================
__global__ __launch_bounds__(320)
void spans_kernel(const float* __restrict__ seq, const int* __restrict__ cspan,
                  const int* __restrict__ aspan, const unsigned char* __restrict__ amask,
                  float* __restrict__ cemb, float* __restrict__ adus)
{
  const int c = blockIdx.x, n = threadIdx.x;
  const float* sq = seq + (size_t)c * L_N * 160;
  auto rep = [&](int i, int j) -> float {
    if (n < 80)       return sq[(size_t)j * 160 + n] - sq[(size_t)(i - 1) * 160 + n];
    else if (n < 160) return sq[(size_t)i * 160 + n] - sq[(size_t)(j + 1) * 160 + n];
    else if (n < 240) return sq[(size_t)(i - 1) * 160 + (n - 160)];
    else              return sq[(size_t)(j + 1) * 160 + 80 + (n - 240)];
  };
  const int ci = cspan[c * 2], cj = cspan[c * 2 + 1];
  cemb[(size_t)c * SPAN + n] = rep(ci, cj);
  for (int a = 0; a < A_N; ++a) {
    int i = aspan[((size_t)c * A_N + a) * 2];
    int j = aspan[((size_t)c * A_N + a) * 2 + 1];
    float v = rep(i, j);
    adus[((size_t)c * A_N + a) * SPAN + n] = amask[c * A_N + a] ? v : 0.0f;
  }
}

// =====================================================================
// GAT per (comment, metapath): masked edge softmax + aggregation + ELU.
// hp rows: [c*32+a][m*768 + k*256 + o]
// =====================================================================
__global__ __launch_bounds__(256)
void gat_kernel(const float* __restrict__ hp,
                const int* isrc, const int* idst, const int* irel, const unsigned char* imask,
                const int* tsrc, const int* tdst, const int* trel, const unsigned char* tmask,
                const float* __restrict__ a_l, const float* __restrict__ a_r,
                const float* __restrict__ b_gat, float* __restrict__ zout)
{
  __shared__ int sse[E_N], sde[E_N];
  __shared__ unsigned char smk[E_N];
  __shared__ float dl[A_N][HEADS], drr[A_N][HEADS];
  __shared__ float esh[E_N][HEADS], exsh[E_N][HEADS];
  __shared__ float msh[A_N][HEADS], dsh[A_N][HEADS];
  const int c = blockIdx.x >> 2, m = blockIdx.x & 3;
  const int tid = threadIdx.x;
  if (tid < E_N) {
    if (m < 2) {
      sse[tid] = isrc[c * E_N + tid]; sde[tid] = idst[c * E_N + tid];
      smk[tid] = (imask[c * E_N + tid] && (irel[c * E_N + tid] == m)) ? 1 : 0;
    } else {
      sse[tid] = tdst[c * E_N + tid]; sde[tid] = tsrc[c * E_N + tid];
      smk[tid] = (tmask[c * E_N + tid] && (trel[c * E_N + tid] == (m - 2))) ? 1 : 0;
    }
  }
  __syncthreads();
  if (tid < A_N * HEADS) {
    const int a = tid / HEADS, k = tid % HEADS;
    const float* hrow = hp + ((size_t)(c * A_N + a)) * 3072 + m * 768 + k * 256;
    const float* al = a_l + (m * HEADS + k) * 256;
    const float* ar = a_r + (m * HEADS + k) * 256;
    float s1 = 0.f, s2 = 0.f;
    for (int j = 0; j < 256; ++j) { float h = hrow[j]; s1 += h * al[j]; s2 += h * ar[j]; }
    dl[a][k] = s1; drr[a][k] = s2;
  }
  __syncthreads();
  if (tid < E_N) {
    for (int k = 0; k < HEADS; ++k) {
      float e = dl[sse[tid]][k] + drr[sde[tid]][k];
      e = (e > 0.f) ? e : 0.2f * e;                 // leaky_relu(0.2)
      if (!smk[tid]) e = -1.0e9f;
      esh[tid][k] = e;
    }
  }
  __syncthreads();
  if (tid < A_N * HEADS) {
    const int a = tid / HEADS, k = tid % HEADS;
    float mm = -1.0e9f;
    for (int e = 0; e < E_N; ++e) if (sde[e] == a) mm = fmaxf(mm, esh[e][k]);
    msh[a][k] = mm;
  }
  __syncthreads();
  if (tid < E_N) {
    for (int k = 0; k < HEADS; ++k)
      exsh[tid][k] = smk[tid] ? __expf(esh[tid][k] - msh[sde[tid]][k]) : 0.0f;
  }
  __syncthreads();
  if (tid < A_N * HEADS) {
    const int a = tid / HEADS, k = tid % HEADS;
    float dn = 0.f;
    for (int e = 0; e < E_N; ++e) if (sde[e] == a) dn += exsh[e][k];
    dsh[a][k] = fmaxf(dn, 1e-9f);
  }
  __syncthreads();
  for (int ch = tid; ch < HEADS * 256; ch += 256) {
    const int k = ch >> 8, o = ch & 255;
    for (int a = 0; a < A_N; ++a) {
      float acc = 0.f;
      for (int e = 0; e < E_N; ++e) {
        if (sde[e] == a && smk[e])
          acc += (exsh[e][k] / dsh[a][k]) *
                 hp[((size_t)(c * A_N + sse[e])) * 3072 + m * 768 + k * 256 + o];
      }
      float v = acc + b_gat[(m * HEADS + k) * 256 + o];
      v = (v > 0.f) ? v : (__expf(v) - 1.0f);       // elu
      zout[(((size_t)(c * 4 + m)) * A_N + a) * 768 + ch] = v;
    }
  }
}

// =====================================================================
// beta = softmax over metapaths of masked mean of tanh(semt)@q_sem;
// zfin = sum_m beta[m] * z[m]
// =====================================================================
__global__ __launch_bounds__(256)
void beta_zfin_kernel(const float* __restrict__ semt, const float* __restrict__ q_sem,
                      const unsigned char* __restrict__ amask,
                      const float* __restrict__ zgat, float* __restrict__ zfin)
{
  __shared__ float wv[4 * A_N];
  __shared__ float sbeta[4];
  const int c = blockIdx.x, tid = threadIdx.x;
  if (tid < 4 * A_N) {
    const int m = tid >> 5, a = tid & 31;
    float s = 0.f;
    if (amask[c * A_N + a]) {
      const float* row = semt + (((size_t)(c * 4 + m)) * A_N + a) * 128;
      for (int j = 0; j < 128; ++j) s += tanhf(row[j]) * q_sem[j];
    }
    wv[tid] = s;
  }
  __syncthreads();
  if (tid == 0) {
    int cnt = 0;
    for (int a = 0; a < A_N; ++a) cnt += amask[c * A_N + a] ? 1 : 0;
    const float d = (float)(cnt > 1 ? cnt : 1);
    float w[4], mm = -1e30f;
    for (int m = 0; m < 4; ++m) {
      float s = 0.f;
      for (int a = 0; a < A_N; ++a) s += wv[m * A_N + a];
      w[m] = s / d; mm = fmaxf(mm, w[m]);
    }
    float den = 0.f;
    for (int m = 0; m < 4; ++m) { w[m] = __expf(w[m] - mm); den += w[m]; }
    for (int m = 0; m < 4; ++m) sbeta[m] = w[m] / den;
  }
  __syncthreads();
  for (int idx = tid; idx < A_N * 768; idx += 256) {
    const int a = idx / 768, j = idx % 768;
    float s = 0.f;
    for (int m = 0; m < 4; ++m)
      s += sbeta[m] * zgat[(((size_t)(c * 4 + m)) * A_N + a) * 768 + j];
    zfin[((size_t)(c * A_N + a)) * 768 + j] = s;
  }
}

// =====================================================================
// Attention pools (ADU pool + two pair pools) + info/cemb concat -> wo_ctx
// =====================================================================
__global__ __launch_bounds__(256)
void pools_kernel(const float* cembbuf, const float* adue,
                  const unsigned char* amask, const unsigned char* lmask,
                  const int* isrc, const int* idst, const int* irel, const unsigned char* imask,
                  const int* tsrc, const int* tdst, const int* trel, const unsigned char* tmask,
                  const float* W_adu1, const float* b_adu1, const float* W_adu2, const float* b_adu2,
                  const float* W_inn1, const float* b_inn1, const float* W_inn2, const float* b_inn2,
                  const float* W_int1, const float* b_int1, const float* W_int2, const float* b_int2,
                  const float* info, float* wo)
{
  __shared__ float cemb_sh[SPAN];
  __shared__ float sh_s[E_N], sh_al[E_N];
  __shared__ int sh_se[E_N], sh_de[E_N], sh_rl[E_N];
  __shared__ unsigned char sh_mk[E_N];
  const int c = blockIdx.x, tid = threadIdx.x;
  for (int i = tid; i < SPAN; i += 256) cemb_sh[i] = cembbuf[(size_t)c * SPAN + i];
  __syncthreads();

  // ---- ADU pool (items=32, feats = [cemb(320), adu(256)]) ----
  if (tid < A_N) {
    float hv[16];
    for (int h = 0; h < 16; ++h) hv[h] = b_adu1[h];
    const float* arow = adue + ((size_t)(c * A_N + tid)) * ADU;
    for (int d = 0; d < SPAN + ADU; ++d) {
      const float x = (d < SPAN) ? cemb_sh[d] : arow[d - SPAN];
      const float* wr = W_adu1 + d * 16;
      for (int h = 0; h < 16; ++h) hv[h] += x * wr[h];
    }
    float s = b_adu2[0];
    for (int h = 0; h < 16; ++h) s += fmaxf(hv[h], 0.f) * W_adu2[h];
    sh_s[tid] = s;
    sh_mk[tid] = (amask[c * A_N + tid] && lmask[c * A_N + tid]) ? 1 : 0;
  }
  __syncthreads();
  if (tid == 0) {
    float mm = -1e30f;
    for (int a = 0; a < A_N; ++a) if (sh_mk[a]) mm = fmaxf(mm, sh_s[a]);
    float den = 0.f;
    for (int a = 0; a < A_N; ++a) { float e = sh_mk[a] ? __expf(sh_s[a] - mm) : 0.f; sh_al[a] = e; den += e; }
    for (int a = 0; a < A_N; ++a) sh_al[a] = (den > 0.f) ? sh_al[a] / den : 0.f;
  }
  __syncthreads();
  if (tid < ADU) {
    float acc = 0.f;
    for (int a = 0; a < A_N; ++a) acc += sh_al[a] * adue[((size_t)(c * A_N + a)) * ADU + tid];
    wo[(size_t)c * DWO + tid] = acc;
  }
  __syncthreads();

  // ---- pair pools: pp=0 inner (isrc->idst), pp=1 inter (tdst->tsrc) ----
  for (int pp = 0; pp < 2; ++pp) {
    const int* se_g = (pp == 0) ? isrc : tdst;
    const int* de_g = (pp == 0) ? idst : tsrc;
    const int* rl_g = (pp == 0) ? irel : trel;
    const unsigned char* mk_g = (pp == 0) ? imask : tmask;
    const float* W1 = (pp == 0) ? W_inn1 : W_int1;
    const float* b1 = (pp == 0) ? b_inn1 : b_int1;
    const float* W2 = (pp == 0) ? W_inn2 : W_int2;
    const float* b2 = (pp == 0) ? b_inn2 : b_int2;
    const int obase = 256 + pp * 514;
    if (tid < E_N) {
      sh_se[tid] = se_g[c * E_N + tid]; sh_de[tid] = de_g[c * E_N + tid];
      sh_rl[tid] = rl_g[c * E_N + tid]; sh_mk[tid] = mk_g[c * E_N + tid] ? 1 : 0;
    }
    __syncthreads();
    if (tid < E_N) {
      float hv[16];
      for (int h = 0; h < 16; ++h) hv[h] = b1[h];
      const float* r1 = adue + ((size_t)(c * A_N + sh_se[tid])) * ADU;
      const float* r2 = adue + ((size_t)(c * A_N + sh_de[tid])) * ADU;
      const float oh0 = (float)sh_rl[tid], oh1 = 1.0f - oh0;
      for (int d = 0; d < SPAN + 2 * ADU + 2; ++d) {
        float x;
        if (d < SPAN)            x = cemb_sh[d];
        else if (d < SPAN + ADU) x = r1[d - SPAN];
        else if (d < SPAN + 2 * ADU) x = r2[d - SPAN - ADU];
        else                     x = (d == SPAN + 2 * ADU) ? oh0 : oh1;
        const float* wr = W1 + d * 16;
        for (int h = 0; h < 16; ++h) hv[h] += x * wr[h];
      }
      float s = b2[0];
      for (int h = 0; h < 16; ++h) s += fmaxf(hv[h], 0.f) * W2[h];
      sh_s[tid] = s;
    }
    __syncthreads();
    if (tid == 0) {
      float mm = -1e30f;
      for (int e = 0; e < E_N; ++e) if (sh_mk[e]) mm = fmaxf(mm, sh_s[e]);
      float den = 0.f;
      for (int e = 0; e < E_N; ++e) { float ex = sh_mk[e] ? __expf(sh_s[e] - mm) : 0.f; sh_al[e] = ex; den += ex; }
      for (int e = 0; e < E_N; ++e) sh_al[e] = (den > 0.f) ? sh_al[e] / den : 0.f;
    }
    __syncthreads();
    for (int o = tid; o < 2 * ADU + 2; o += 256) {
      float acc = 0.f;
      for (int e = 0; e < E_N; ++e) {
        float v;
        if (o < ADU)          v = adue[((size_t)(c * A_N + sh_se[e])) * ADU + o];
        else if (o < 2 * ADU) v = adue[((size_t)(c * A_N + sh_de[e])) * ADU + (o - ADU)];
        else if (o == 2 * ADU) v = (float)sh_rl[e];
        else                  v = 1.0f - (float)sh_rl[e];
        acc += sh_al[e] * v;
      }
      wo[(size_t)c * DWO + obase + o] = acc;
    }
    __syncthreads();
  }
  if (tid < 4) wo[(size_t)c * DWO + 1284 + tid] = info[c * 4 + tid];
  for (int i = tid; i < SPAN; i += 256) wo[(size_t)c * DWO + 1288 + i] = cemb_sh[i];
}

// =====================================================================
// Context LSTM over 64 comments (single block, 800 threads = 25 waves).
// =====================================================================
__global__ __launch_bounds__(800)
void ctx_lstm(const float* __restrict__ Zc, const float* __restrict__ Whh,
              float* __restrict__ outh)
{
  __shared__ float sh_h[CTX];
  __shared__ float sh_z[4 * CTX];
  const int tid = threadIdx.x;
  if (tid < CTX) sh_h[tid] = 0.f;
  float cc = 0.f;
  __syncthreads();
  for (int t = 0; t < C_N; ++t) {
    float z = Zc[(size_t)t * 4 * CTX + tid];
    const float* wr = Whh + (size_t)tid * CTX;
    for (int k = 0; k < CTX; ++k) z += sh_h[k] * wr[k];
    __syncthreads();
    sh_z[tid] = z;
    __syncthreads();
    if (tid < CTX) {
      float iv = sigf(sh_z[tid]);
      float fv = sigf(sh_z[CTX + tid]);
      float gv = tanhf(sh_z[2 * CTX + tid]);
      float ov = sigf(sh_z[3 * CTX + tid]);
      cc = fv * cc + iv * gv;
      float h = ov * tanhf(cc);
      sh_h[tid] = h;
      outh[(size_t)t * CTX + tid] = h;
    }
    __syncthreads();
  }
}

__global__ __launch_bounds__(256)
void final_kernel(const float* ctxh, const float* wo, float* out)
{
  const int c = blockIdx.x, tid = threadIdx.x;
  for (int j = tid; j < DOUT; j += 256)
    out[(size_t)c * DOUT + j] =
        (j < CTX) ? ctxh[(size_t)c * CTX + j] : wo[(size_t)c * DWO + (j - CTX)];
}

// =====================================================================
extern "C" void kernel_launch(void* const* d_in, const int* in_sizes, int n_in,
                              void* d_out, int out_size, void* d_ws, size_t ws_size,
                              hipStream_t stream)
{
  (void)in_sizes; (void)n_in; (void)out_size; (void)ws_size;
  // setup_inputs() dict order:
  const float* token = (const float*)d_in[0];
  const float* info  = (const float*)d_in[1];
  const int* cspan   = (const int*)d_in[2];
  const int* aspan   = (const int*)d_in[3];
  const unsigned char* amask = (const unsigned char*)d_in[4];
  const unsigned char* lmask = (const unsigned char*)d_in[5];
  const int* isrc = (const int*)d_in[6];  const int* idst = (const int*)d_in[7];
  const int* irel = (const int*)d_in[8];  const unsigned char* imask = (const unsigned char*)d_in[9];
  const int* tsrc = (const int*)d_in[10]; const int* tdst = (const int*)d_in[11];
  const int* trel = (const int*)d_in[12]; const unsigned char* tmask = (const unsigned char*)d_in[13];
  const float* Wih_f = (const float*)d_in[14]; const float* Whh_f = (const float*)d_in[15]; const float* b_f = (const float*)d_in[16];
  const float* Wih_b = (const float*)d_in[17]; const float* Whh_b = (const float*)d_in[18]; const float* b_b = (const float*)d_in[19];
  const float* W_gat = (const float*)d_in[20]; const float* a_l = (const float*)d_in[21];
  const float* a_r   = (const float*)d_in[22]; const float* b_gat = (const float*)d_in[23];
  const float* W_sem = (const float*)d_in[24]; const float* b_sem = (const float*)d_in[25]; const float* q_sem = (const float*)d_in[26];
  const float* W_pred = (const float*)d_in[27]; const float* b_pred = (const float*)d_in[28];
  const float* W_adu1 = (const float*)d_in[29]; const float* b_adu1 = (const float*)d_in[30];
  const float* W_adu2 = (const float*)d_in[31]; const float* b_adu2 = (const float*)d_in[32];
  const float* W_inn1 = (const float*)d_in[33]; const float* b_inn1 = (const float*)d_in[34];
  const float* W_inn2 = (const float*)d_in[35]; const float* b_inn2 = (const float*)d_in[36];
  const float* W_int1 = (const float*)d_in[37]; const float* b_int1 = (const float*)d_in[38];
  const float* W_int2 = (const float*)d_in[39]; const float* b_int2 = (const float*)d_in[40];
  const float* Wih_c = (const float*)d_in[41]; const float* Whh_c = (const float*)d_in[42]; const float* b_c = (const float*)d_in[43];

  // ---- workspace layout (bytes), with aliasing reuse after the LSTM ----
  char* ws = (char*)d_ws;
  const size_t SEQ_OFF    = 0;
  const size_t ADUS_OFF   = SEQ_OFF   + (size_t)C_N * L_N * 160 * 4;   // 41.9 MB
  const size_t CEMB_OFF   = ADUS_OFF  + (size_t)C_N * A_N * SPAN * 4;
  const size_t WO_OFF     = CEMB_OFF  + (size_t)C_N * SPAN * 4;
  const size_t CTXPRE_OFF = WO_OFF    + (size_t)C_N * DWO * 4;
  const size_t CTXH_OFF   = CTXPRE_OFF + (size_t)C_N * 4 * CTX * 4;
  const size_t ZF_OFF     = CTXH_OFF  + (size_t)C_N * CTX * 4;
  const size_t ZB_OFF     = ZF_OFF    + (size_t)C_N * L_N * GATE * 4;  // 83.9 MB each
  // aliases (ZF/ZB regions are dead after lstm_seq):
  const size_t HP_OFF   = ZF_OFF;
  const size_t ZGAT_OFF = ZF_OFF + (size_t)C_N * A_N * 3072 * 4;       // +25.2 MB
  const size_t SEMT_OFF = ZGAT_OFF + (size_t)C_N * 4 * A_N * 768 * 4;  // +25.2 MB
  const size_t ZFIN_OFF = ZB_OFF;
  const size_t ADUE_OFF = ZB_OFF + (size_t)C_N * A_N * 768 * 4;        // +6.3 MB

  float* SEQ    = (float*)(ws + SEQ_OFF);
  float* ADUS   = (float*)(ws + ADUS_OFF);
  float* CEMB   = (float*)(ws + CEMB_OFF);
  float* WO     = (float*)(ws + WO_OFF);
  float* CTXPRE = (float*)(ws + CTXPRE_OFF);
  float* CTXH   = (float*)(ws + CTXH_OFF);
  float* ZF     = (float*)(ws + ZF_OFF);
  float* ZB     = (float*)(ws + ZB_OFF);
  float* HP     = (float*)(ws + HP_OFF);
  float* ZGAT   = (float*)(ws + ZGAT_OFF);
  float* SEMT   = (float*)(ws + SEMT_OFF);
  float* ZFIN   = (float*)(ws + ZFIN_OFF);
  float* ADUE   = (float*)(ws + ADUE_OFF);

  auto nblk = [](int n) { return (n + 63) / 64; };

  // 1) LSTM input projections (the FLOP-dominant GEMMs): 65536x320x768 each
  gemm_wmma<1><<<dim3(nblk(GATE), (C_N * L_N) / 64), 256, 0, stream>>>(
      token, Wih_f, b_f, ZF, C_N * L_N, GATE, EMB);
  gemm_wmma<1><<<dim3(nblk(GATE), (C_N * L_N) / 64), 256, 0, stream>>>(
      token, Wih_b, b_b, ZB, C_N * L_N, GATE, EMB);

  // 2) Recurrences (fwd + bwd), 64 independent sequences each
  lstm_seq<<<C_N, 320, 0, stream>>>(ZF, Whh_f, SEQ, 0, 0);
  lstm_seq<<<C_N, 320, 0, stream>>>(ZB, Whh_b, SEQ, 1, H_L);

  // 3) Span reps
  spans_kernel<<<C_N, 320, 0, stream>>>(SEQ, cspan, aspan, amask, CEMB, ADUS);

  // 4) GAT head projection fused over 4 metapaths x 3 heads: 2048x3072x320
  gemm_wmma<2><<<dim3(nblk(3072), (C_N * A_N) / 64), 256, 0, stream>>>(
      ADUS, W_gat, nullptr, HP, C_N * A_N, 3072, SPAN);

  // 5) GAT attention + aggregation per (comment, metapath)
  gat_kernel<<<C_N * 4, 256, 0, stream>>>(HP, isrc, idst, irel, imask,
                                          tsrc, tdst, trel, tmask,
                                          a_l, a_r, b_gat, ZGAT);

  // 6) Semantic projection 8192x128x768, then beta-weighted mix
  gemm_wmma<0><<<dim3(nblk(128), (C_N * 4 * A_N) / 64), 256, 0, stream>>>(
      ZGAT, W_sem, b_sem, SEMT, C_N * 4 * A_N, 128, 768);
  beta_zfin_kernel<<<C_N, 256, 0, stream>>>(SEMT, q_sem, amask, ZGAT, ZFIN);

  // 7) Predicate projection 2048x256x768
  gemm_wmma<0><<<dim3(nblk(ADU), (C_N * A_N) / 64), 256, 0, stream>>>(
      ZFIN, W_pred, b_pred, ADUE, C_N * A_N, ADU, 768);

  // 8) Attention pools + concat -> wo_ctx
  pools_kernel<<<C_N, 256, 0, stream>>>(CEMB, ADUE, amask, lmask,
                                        isrc, idst, irel, imask,
                                        tsrc, tdst, trel, tmask,
                                        W_adu1, b_adu1, W_adu2, b_adu2,
                                        W_inn1, b_inn1, W_inn2, b_inn2,
                                        W_int1, b_int1, W_int2, b_int2,
                                        info, WO);

  // 9) Context LSTM: input projection 64x800x1608 (K tail handled), then scan
  gemm_wmma<1><<<dim3(nblk(4 * CTX), 1), 256, 0, stream>>>(
      WO, Wih_c, b_c, CTXPRE, C_N, 4 * CTX, DWO);
  ctx_lstm<<<1, 800, 0, stream>>>(CTXPRE, Whh_c, CTXH);

  // 10) Final concat [ctx(200), wo_ctx(1608)]
  final_kernel<<<C_N, 256, 0, stream>>>(CTXH, WO, (float*)d_out);
}
